// VectorQuantizer_11570641896079
// MI455X (gfx1250) — matrix operand order repaired and checked by reading, not compiled, not run on hardware
//
#include <hip/hip_runtime.h>
#include <hip/hip_bf16.h>

typedef _Float16 v8h  __attribute__((ext_vector_type(8)));
typedef _Float16 v16h __attribute__((ext_vector_type(16)));
typedef float    v8f  __attribute__((ext_vector_type(8)));

#define NUM_CODES 512
#define CODE_DIM  64
#define N_ROWS    (32 * 64 * 64)          // 131072 latent vectors
#define CB_LDS_STRIDE 72                  // halfs: 144B row stride -> conflict-free ds_load_b128

// ---------------- ws layout (bytes) ----------------
// [0, 65536)        : codebook f16, row stride 64 halfs
// [65536, 67584)    : c2[512] f32
// [67584, 69632)    : hist[512] u32
// [69632, 69636)    : mse accumulator f32
#define WS_CB16 0
#define WS_C2   65536
#define WS_HIST 67584
#define WS_MSE  69632

// ---------------------------------------------------------------------------
// Kernel 0: codebook f32 -> f16, c2 = ||c||^2, zero hist + mse (every launch)
// ---------------------------------------------------------------------------
__global__ __launch_bounds__(256) void vq_prep_kernel(
    const float* __restrict__ codebook, _Float16* __restrict__ cb16,
    float* __restrict__ c2, unsigned* __restrict__ g_hist, float* __restrict__ mse)
{
  int t = blockIdx.x * blockDim.x + threadIdx.x;   // 0..511
  if (t < NUM_CODES) {
    float s = 0.0f;
#pragma unroll
    for (int i = 0; i < CODE_DIM; ++i) {
      float v = codebook[t * CODE_DIM + i];
      s = fmaf(v, v, s);
      cb16[t * CODE_DIM + i] = (_Float16)v;
    }
    c2[t] = s;
    g_hist[t] = 0u;
    if (t == 0) *mse = 0.0f;
  }
}

// ---------------------------------------------------------------------------
// Kernel 1 (fused): WMMA argmin codes + histogram + quantized_st + MSE.
// Block = 256 thr = 8 wave32 = 128 rows. Each wave: 16 rows x all 512 codes.
// score = c2 - 2*(x . c)   (||x||^2 dropped: constant per row for argmin)
// ---------------------------------------------------------------------------
__global__ __launch_bounds__(256) void vq_main_kernel(
    const float* __restrict__ latents, const float* __restrict__ codebook,
    const _Float16* __restrict__ cb16, const float* __restrict__ c2,
    unsigned* __restrict__ g_hist, float* __restrict__ codes_out,
    float* __restrict__ qst, float* __restrict__ mse)
{
  __shared__ _Float16 cbL[NUM_CODES * CB_LDS_STRIDE];  // 73728 B
  __shared__ float    c2L[NUM_CODES];                  //  2048 B
  __shared__ unsigned histL[NUM_CODES];                //  2048 B
  __shared__ int      codesL[128];                     //   512 B
  __shared__ float    red[256];                        //  1024 B

  const int tid = threadIdx.x;

  // init hist + c2, stage f16 codebook into LDS (padded stride)
  histL[tid]       = 0u;
  histL[tid + 256] = 0u;
  c2L[tid]         = c2[tid];
  c2L[tid + 256]   = c2[tid + 256];
#pragma unroll
  for (int rr = 0; rr < 2; ++rr) {
    int r = tid + rr * 256;
    const uint4* src = (const uint4*)(cb16 + r * CODE_DIM);   // 128B row
    uint4*       dst = (uint4*)(cbL + r * CB_LDS_STRIDE);
#pragma unroll
    for (int i = 0; i < 8; ++i) dst[i] = src[i];
  }
  __syncthreads();

  const int lane = tid & 31;
  const int wv   = tid >> 5;
  const int half = lane >> 4;        // 0: lanes 0-15, 1: lanes 16-31
  const int lr   = lane & 15;
  const int row_base = blockIdx.x * 128 + wv * 16;

  // --- A fragments: ISA 16-bit A 16x32 layout.
  // lane (half,lr): row = row_base+lr, chunk0 K in {8h..8h+7} U {8h+16..8h+23}
  const float4* lr4 = (const float4*)(latents + (size_t)(row_base + lr) * CODE_DIM);
  const int bi = half * 2;
  float4 fA0 = lr4[bi],      fA1 = lr4[bi + 1];    // K 8h..8h+7
  float4 fB0 = lr4[bi + 4],  fB1 = lr4[bi + 5];    // K 8h+16..8h+23
  float4 fC0 = lr4[bi + 8],  fC1 = lr4[bi + 9];    // +32
  float4 fD0 = lr4[bi + 12], fD1 = lr4[bi + 13];
  v16h a0, a1;
  a0[0]=(_Float16)fA0.x; a0[1]=(_Float16)fA0.y; a0[2]=(_Float16)fA0.z; a0[3]=(_Float16)fA0.w;
  a0[4]=(_Float16)fA1.x; a0[5]=(_Float16)fA1.y; a0[6]=(_Float16)fA1.z; a0[7]=(_Float16)fA1.w;
  a0[8]=(_Float16)fB0.x; a0[9]=(_Float16)fB0.y; a0[10]=(_Float16)fB0.z; a0[11]=(_Float16)fB0.w;
  a0[12]=(_Float16)fB1.x; a0[13]=(_Float16)fB1.y; a0[14]=(_Float16)fB1.z; a0[15]=(_Float16)fB1.w;
  a1[0]=(_Float16)fC0.x; a1[1]=(_Float16)fC0.y; a1[2]=(_Float16)fC0.z; a1[3]=(_Float16)fC0.w;
  a1[4]=(_Float16)fC1.x; a1[5]=(_Float16)fC1.y; a1[6]=(_Float16)fC1.z; a1[7]=(_Float16)fC1.w;
  a1[8]=(_Float16)fD0.x; a1[9]=(_Float16)fD0.y; a1[10]=(_Float16)fD0.z; a1[11]=(_Float16)fD0.w;
  a1[12]=(_Float16)fD1.x; a1[13]=(_Float16)fD1.y; a1[14]=(_Float16)fD1.z; a1[15]=(_Float16)fD1.w;

  float bestd[8];
  int   besti[8];
#pragma unroll
  for (int v = 0; v < 8; ++v) { bestd[v] = 3.4e38f; besti[v] = 0; }

  // B fragment loader: ISA 16-bit B 32x16: lane holds col n, K {8h..8h+7} U {8h+16..8h+23}
  const _Float16* browBase = cbL + lr * CB_LDS_STRIDE + half * 8;

  // software pipeline: load tile ct+1 while computing tile ct
  v8h p0, p1, q0, q1, np0, np1, nq0, nq1;
  float c2n, nc2;
  {
    const _Float16* brow = browBase;                       // n = lr (ct = 0)
    p0 = *(const v8h*)(brow);      p1 = *(const v8h*)(brow + 16);
    q0 = *(const v8h*)(brow + 32); q1 = *(const v8h*)(brow + 48);
    c2n = c2L[lr];
  }

  for (int ct = 0; ct < 32; ++ct) {
    if (ct < 31) {   // prefetch next tile's B fragment + c2 (overlaps WMMA below)
      const int nn = (ct + 1) * 16 + lr;
      const _Float16* brow = browBase + (ct + 1) * 16 * CB_LDS_STRIDE;
      np0 = *(const v8h*)(brow);      np1 = *(const v8h*)(brow + 16);
      nq0 = *(const v8h*)(brow + 32); nq1 = *(const v8h*)(brow + 48);
      nc2 = c2L[nn];
    }

    v16h b0, b1;
#pragma unroll
    for (int i = 0; i < 8; ++i) { b0[i] = p0[i]; b0[8 + i] = p1[i]; b1[i] = q0[i]; b1[8 + i] = q1[i]; }

    v8f acc = {};
    acc = __builtin_amdgcn_wmma_f32_16x16x32_f16(false, a0, false, b0, (short)0, acc, false, false);
    acc = __builtin_amdgcn_wmma_f32_16x16x32_f16(false, a1, false, b1, (short)0, acc, false, false);

    const int n = ct * 16 + lr;
#pragma unroll
    for (int v = 0; v < 8; ++v) {
      float s = fmaf(-2.0f, acc[v], c2n);
      bool take = s < bestd[v];                     // strict <: keeps lowest index
      bestd[v] = take ? s : bestd[v];
      besti[v] = take ? n : besti[v];
    }

    p0 = np0; p1 = np1; q0 = nq0; q1 = nq1; c2n = nc2;
  }

  // --- branchless min-reduce across the 16 lanes of each half (ties -> lower idx)
#pragma unroll
  for (int off = 8; off >= 1; off >>= 1) {
#pragma unroll
    for (int v = 0; v < 8; ++v) {
      float od = __shfl_xor(bestd[v], off, 16);
      int   oi = __shfl_xor(besti[v], off, 16);
      bool take = (od < bestd[v]) || (od == bestd[v] && oi < besti[v]);
      bestd[v] = take ? od : bestd[v];
      besti[v] = take ? oi : besti[v];
    }
  }

  if (lr == 0) {
#pragma unroll
    for (int v = 0; v < 8; ++v) {
      int mloc = half * 8 + v;                    // C/D layout: m = v + 8*half
      codesL[wv * 16 + mloc] = besti[v];
      codes_out[row_base + mloc] = (float)besti[v];
      atomicAdd(&histL[besti[v]], 1u);
    }
  }
  __syncthreads();

  // --- fused quantize + MSE: thread handles half a row (32 floats) ---
  {
    const int rloc = tid >> 1;                    // 0..127
    const int seg  = tid & 1;                     // 0/1
    const int rowg = blockIdx.x * 128 + rloc;
    const int code = codesL[rloc];
    const float4* l4 = (const float4*)(latents  + (size_t)rowg * CODE_DIM + seg * 32);
    const float4* q4 = (const float4*)(codebook + (size_t)code * CODE_DIM + seg * 32);
    float4*       o4 = (float4*)(qst + (size_t)rowg * CODE_DIM + seg * 32);
    float s = 0.0f;
#pragma unroll
    for (int i = 0; i < 8; ++i) {
      float4 l = l4[i], q = q4[i], o;
      o.x = l.x + (q.x - l.x); o.y = l.y + (q.y - l.y);
      o.z = l.z + (q.z - l.z); o.w = l.w + (q.w - l.w);
      float dx = l.x - q.x, dy = l.y - q.y, dz = l.z - q.z, dw = l.w - q.w;
      s += dx * dx + dy * dy + dz * dz + dw * dw;
      o4[i] = o;
    }
    red[tid] = s;
  }
  __syncthreads();
  for (int o = 128; o > 0; o >>= 1) {
    if (tid < o) red[tid] += red[tid + o];
    __syncthreads();
  }
  if (tid == 0) atomicAdd(mse, red[0]);

  // flush histogram
  if (histL[tid])       atomicAdd(&g_hist[tid],       histL[tid]);
  if (histL[tid + 256]) atomicAdd(&g_hist[tid + 256], histL[tid + 256]);
}

// ---------------------------------------------------------------------------
// Kernel 2: losses + perplexity
// ---------------------------------------------------------------------------
__global__ __launch_bounds__(512) void vq_final_kernel(
    const unsigned* __restrict__ g_hist, const float* __restrict__ mse,
    float* __restrict__ scalars)
{
  __shared__ float red[512];
  const int t = threadIdx.x;
  float p = (float)g_hist[t] / (float)N_ROWS;
  red[t] = -p * logf(p + 1e-10f);
  __syncthreads();
  for (int o = 256; o > 0; o >>= 1) {
    if (t < o) red[t] += red[t + o];
    __syncthreads();
  }
  if (t == 0) {
    float mean = (*mse) / (float)((size_t)N_ROWS * CODE_DIM);
    scalars[0] = mean * 0.25f;   // commitment * COMMITMENT_COST
    scalars[1] = mean;           // codebook_loss (same value)
    scalars[2] = expf(red[0]);   // perplexity
  }
}

// ---------------------------------------------------------------------------
extern "C" void kernel_launch(void* const* d_in, const int* in_sizes, int n_in,
                              void* d_out, int out_size, void* d_ws, size_t ws_size,
                              hipStream_t stream) {
  const float* latents  = (const float*)d_in[0];
  const float* codebook = (const float*)d_in[1];

  float* out     = (float*)d_out;
  float* qst     = out;                                  // [N_ROWS*64]
  float* codes_f = out + (size_t)N_ROWS * CODE_DIM;      // [N_ROWS]
  float* scalars = codes_f + N_ROWS;                     // [3]

  char*     ws   = (char*)d_ws;
  _Float16* cb16 = (_Float16*)(ws + WS_CB16);
  float*    c2   = (float*)(ws + WS_C2);
  unsigned* hist = (unsigned*)(ws + WS_HIST);
  float*    mse  = (float*)(ws + WS_MSE);

  vq_prep_kernel<<<2, 256, 0, stream>>>(codebook, cb16, c2, hist, mse);
  vq_main_kernel<<<N_ROWS / 128, 256, 0, stream>>>(latents, codebook, cb16, c2,
                                                   hist, codes_f, qst, mse);
  vq_final_kernel<<<1, 512, 0, stream>>>(hist, mse, scalars);
}
